// Adj_3779571220955
// MI455X (gfx1250) — matrix operand order repaired and checked by reading, not compiled, not run on hardware
//
#include <hip/hip_runtime.h>

typedef __attribute__((ext_vector_type(8)))  float   v8f;
typedef __attribute__((ext_vector_type(8)))  __bf16  v8bf;
typedef __attribute__((ext_vector_type(16))) __bf16  v16bf;

#define BATCH 64
#define NROW  512
#define CDIM  1024
#define TILE  128
#define KT    32
#define LSTR  40      // KT + 8 pad (bf16 elements) -> conflict-free b128 frag reads
#define NEIGH 32

// ---- helpers ---------------------------------------------------------------

// Split a float4 into bf16 hi (truncate) and bf16 lo (residual, truncate),
// packed as 4x u16 each, stored as one 8-byte LDS store per tile.
__device__ __forceinline__ void split_store(unsigned short* dh, unsigned short* dl, float4 v) {
    unsigned int u0 = __float_as_uint(v.x), u1 = __float_as_uint(v.y),
                 u2 = __float_as_uint(v.z), u3 = __float_as_uint(v.w);
    float h0 = __uint_as_float(u0 & 0xFFFF0000u);
    float h1 = __uint_as_float(u1 & 0xFFFF0000u);
    float h2 = __uint_as_float(u2 & 0xFFFF0000u);
    float h3 = __uint_as_float(u3 & 0xFFFF0000u);
    unsigned int l0 = __float_as_uint(v.x - h0);
    unsigned int l1 = __float_as_uint(v.y - h1);
    unsigned int l2 = __float_as_uint(v.z - h2);
    unsigned int l3 = __float_as_uint(v.w - h3);
    uint2 hp, lp;
    hp.x = (u0 >> 16) | (u1 & 0xFFFF0000u);   // elem0 in [15:0] (K even low half)
    hp.y = (u2 >> 16) | (u3 & 0xFFFF0000u);
    lp.x = (l0 >> 16) | (l1 & 0xFFFF0000u);
    lp.y = (l2 >> 16) | (l3 & 0xFFFF0000u);
    *(uint2*)dh = hp;
    *(uint2*)dl = lp;
}

// Build a v16bf fragment from two 16-byte LDS chunks.
__device__ __forceinline__ v16bf frag16(const unsigned short* t, int o0, int o1) {
    v8bf c0 = *(const v8bf*)(t + o0);
    v8bf c1 = *(const v8bf*)(t + o1);
    return __builtin_shufflevector(c0, c1, 0,1,2,3,4,5,6,7,8,9,10,11,12,13,14,15);
}

// ---- kernel 1: S = X X^T via bf16x3 WMMA -----------------------------------

__global__ __launch_bounds__(256)
void adj_gemm_kernel(const float* __restrict__ x, float* __restrict__ out) {
    __shared__ unsigned short sAh[TILE * LSTR];
    __shared__ unsigned short sAl[TILE * LSTR];
    __shared__ unsigned short sBh[TILE * LSTR];
    __shared__ unsigned short sBl[TILE * LSTR];

    const int b    = blockIdx.z;
    const int i0   = blockIdx.y * TILE;
    const int j0   = blockIdx.x * TILE;
    const int tid  = threadIdx.x;
    const int lane = tid & 31;
    const int wv   = tid >> 5;           // 8 waves
    const int wi   = (wv >> 2) * 64;     // wave tile: 64 x 32
    const int wj   = (wv & 3) * 32;
    const float* xb = x + (size_t)b * NROW * CDIM;

    v8f acc[4][2];
    v8f zero = {0.f,0.f,0.f,0.f,0.f,0.f,0.f,0.f};
    #pragma unroll
    for (int m = 0; m < 4; ++m)
        #pragma unroll
        for (int n = 0; n < 2; ++n)
            acc[m][n] = zero;

    const int lrow = tid >> 3;   // 0..31
    const int lc   = tid & 7;    // float4 column 0..7 (32 floats = KT)

    for (int k0 = 0; k0 < CDIM; k0 += KT) {
        #pragma unroll
        for (int q = 0; q < 4; ++q) {
            int r = lrow + q * 32;
            float4 av = ((const float4*)(xb + (size_t)(i0 + r) * CDIM + k0))[lc];
            float4 bv = ((const float4*)(xb + (size_t)(j0 + r) * CDIM + k0))[lc];
            int so = r * LSTR + lc * 4;
            split_store(&sAh[so], &sAl[so], av);
            split_store(&sBh[so], &sBl[so], bv);
        }
        __syncthreads();

        const int fr = lane & 15;
        const int fh = lane >> 4;
        // A fragment (16-bit, 16x32): chunk0 K=8h..8h+7, chunk1 K=16+8h..16+8h+7
        // B fragment (16-bit, 32x16): chunk0 K=8h..8h+7, chunk1 K=16+8h..16+8h+7
        v16bf ah[4], al[4], bh[2], bl[2];
        #pragma unroll
        for (int m = 0; m < 4; ++m) {
            int base = (wi + m * 16 + fr) * LSTR;
            ah[m] = frag16(sAh + base, 8 * fh, 16 + 8 * fh);
            al[m] = frag16(sAl + base, 8 * fh, 16 + 8 * fh);
        }
        #pragma unroll
        for (int n = 0; n < 2; ++n) {
            int base = (wj + n * 16 + fr) * LSTR;
            bh[n] = frag16(sBh + base, 8 * fh, 16 + 8 * fh);
            bl[n] = frag16(sBl + base, 8 * fh, 16 + 8 * fh);
        }
        #pragma unroll
        for (int m = 0; m < 4; ++m) {
            #pragma unroll
            for (int n = 0; n < 2; ++n) {
                acc[m][n] = __builtin_amdgcn_wmma_f32_16x16x32_bf16(
                    false, ah[m], false, bh[n], (short)0, acc[m][n], false, false);
                acc[m][n] = __builtin_amdgcn_wmma_f32_16x16x32_bf16(
                    false, ah[m], false, bl[n], (short)0, acc[m][n], false, false);
                acc[m][n] = __builtin_amdgcn_wmma_f32_16x16x32_bf16(
                    false, al[m], false, bh[n], (short)0, acc[m][n], false, false);
            }
        }
        __syncthreads();
    }

    // C/D layout: VGPR v, lanes 0-15 -> M=v, lanes 16-31 -> M=v+8; N = lane%16
    float* outb = out + (size_t)b * NROW * NROW;
    const int half = lane >> 4;
    const int col  = lane & 15;
    #pragma unroll
    for (int m = 0; m < 4; ++m)
        #pragma unroll
        for (int n = 0; n < 2; ++n)
            #pragma unroll
            for (int v = 0; v < 8; ++v) {
                int gi = i0 + wi + m * 16 + v + 8 * half;
                int gj = j0 + wj + n * 16 + col;
                outb[(size_t)gi * NROW + gj] = acc[m][n][v];
            }
}

// ---- kernel 2: per-row top-32 threshold + degree ---------------------------

__global__ __launch_bounds__(256)
void adj_topk_kernel(const float* __restrict__ S,
                     float* __restrict__ thr_out, float* __restrict__ dinv_out) {
    const int lane = threadIdx.x & 31;
    const int row  = blockIdx.x * 8 + (threadIdx.x >> 5);
    const float* srow = S + (size_t)row * NROW;

    float v[16];
    #pragma unroll
    for (int t = 0; t < 16; ++t) v[t] = srow[t * 32 + lane];

    float thr = 0.f;
    for (int it = 0; it < NEIGH; ++it) {
        float m = -INFINITY; int mi = 0;
        #pragma unroll
        for (int t = 0; t < 16; ++t)
            if (v[t] > m) { m = v[t]; mi = t; }
        float wm = m;
        #pragma unroll
        for (int off = 16; off > 0; off >>= 1)
            wm = fmaxf(wm, __shfl_xor(wm, off, 32));
        unsigned long long bal = __ballot(m == wm);
        int winner = __ffsll(bal) - 1;
        if (lane == winner) v[mi] = -INFINITY;   // remove exactly one instance
        thr = wm;                                // last removed == 32nd largest
    }

    // degree = 32 removed + remaining ties equal to threshold (matches >=)
    int cnt = 0;
    #pragma unroll
    for (int t = 0; t < 16; ++t) cnt += (v[t] == thr) ? 1 : 0;
    #pragma unroll
    for (int off = 16; off > 0; off >>= 1)
        cnt += __shfl_xor(cnt, off, 32);

    if (lane == 0) {
        thr_out[row]  = thr;
        dinv_out[row] = rsqrtf((float)(NEIGH + cnt));
    }
}

// ---- kernel 3: in-place normalized adjacency -------------------------------

__global__ __launch_bounds__(256)
void adj_norm_kernel(float* __restrict__ S,
                     const float* __restrict__ thr, const float* __restrict__ dinv) {
    const int idx4 = blockIdx.x * 256 + threadIdx.x;       // over 64*512*128 float4s
    const int b    = idx4 / (NROW * (NROW / 4));
    const int rem  = idx4 % (NROW * (NROW / 4));
    const int i    = rem / (NROW / 4);
    const int j4   = rem % (NROW / 4);

    float4 s  = ((const float4*)S)[idx4];
    float  t  = thr[b * NROW + i];
    float  di = dinv[b * NROW + i];
    float4 dj = ((const float4*)dinv)[b * (NROW / 4) + j4];

    float4 o;
    o.x = (s.x >= t) ? di * dj.x : 0.f;
    o.y = (s.y >= t) ? di * dj.y : 0.f;
    o.z = (s.z >= t) ? di * dj.z : 0.f;
    o.w = (s.w >= t) ? di * dj.w : 0.f;
    ((float4*)S)[idx4] = o;
}

// ---- launcher ---------------------------------------------------------------

extern "C" void kernel_launch(void* const* d_in, const int* in_sizes, int n_in,
                              void* d_out, int out_size, void* d_ws, size_t ws_size,
                              hipStream_t stream) {
    (void)in_sizes; (void)n_in; (void)out_size; (void)ws_size;
    const float* x  = (const float*)d_in[0];
    float* out      = (float*)d_out;
    float* thr      = (float*)d_ws;                 // 64*512 floats
    float* dinv     = thr + BATCH * NROW;           // 64*512 floats (256 KB total)

    dim3 g1(NROW / TILE, NROW / TILE, BATCH);       // 4 x 4 x 64
    adj_gemm_kernel<<<g1, 256, 0, stream>>>(x, out);

    adj_topk_kernel<<<(BATCH * NROW) / 8, 256, 0, stream>>>(out, thr, dinv);

    adj_norm_kernel<<<(BATCH * NROW * (NROW / 4)) / 256, 256, 0, stream>>>(out, thr, dinv);
}
